// FusedMHA_90056874262680
// MI455X (gfx1250) — matrix operand order repaired and checked by reading, not compiled, not run on hardware
//
#include <hip/hip_runtime.h>

// Problem constants (B,S,D,H from reference)
#define B_   4
#define S_   2048
#define D_   1024
#define H_   16
#define HD_  64
#define N3_  3072
#define SCALE_ 0.125f   // 1/sqrt(64)

typedef __bf16 bf16_t;
typedef __bf16 v16bf __attribute__((ext_vector_type(16)));
typedef float  v8f   __attribute__((ext_vector_type(8)));

struct alignas(16) Chunk8 { unsigned int w[4]; };   // 8 bf16 = 16 bytes

union Frag16 {           // one WMMA 16-bit A/B operand: 16 bf16 per lane
  v16bf  v;
  Chunk8 q[2];
};

__device__ __forceinline__ unsigned short bfbits(float f) {
  unsigned int u = __builtin_bit_cast(unsigned int, f);
  u += 0x7FFFu + ((u >> 16) & 1u);           // round-to-nearest-even
  return (unsigned short)(u >> 16);
}
__device__ __forceinline__ bf16_t f2bf(float f) {
  unsigned short h = bfbits(f);
  return __builtin_bit_cast(bf16_t, h);
}
__device__ __forceinline__ v8f vzero8() {
  v8f z = {0.f, 0.f, 0.f, 0.f, 0.f, 0.f, 0.f, 0.f};
  return z;
}
__device__ __forceinline__ v8f wmma_bf16(const Frag16& a, const Frag16& b, v8f c) {
  return __builtin_amdgcn_wmma_f32_16x16x32_bf16(false, a.v, false, b.v,
                                                 (short)0, c, false, false);
}

// ---------------------------------------------------------------------------
// Precision conversion / weight transpose pre-passes
// ---------------------------------------------------------------------------
__global__ void cvt_bf16(const float* __restrict__ in, bf16_t* __restrict__ out, int n) {
  int i = blockIdx.x * blockDim.x + threadIdx.x;
  if (i < n) out[i] = f2bf(in[i]);
}

// in: [K][N] f32  ->  out: [N][K] bf16 (so GEMM B-fragments read contiguous K)
__global__ void transpose_bf16(const float* __restrict__ in, bf16_t* __restrict__ out,
                               int K, int N) {
  int i = blockIdx.x * blockDim.x + threadIdx.x;
  if (i < K * N) {
    int k = i / N, n = i % N;
    out[n * K + k] = f2bf(in[i]);
  }
}

// ---------------------------------------------------------------------------
// GEMM 1: qkv = x @ w_qkv  (M=8192, N=3072, K=1024), scatter into
//   Q[b,h,s,hd] (pre-scaled), K[b,h,s,hd], V^T[b,h,hd,s]  -- all bf16
// Block: 256 thr (8 waves), tile 128x64, per-wave 32x32 (2x2 WMMA)
// Low register pressure => backend software-pipelines K loop itself.
// ---------------------------------------------------------------------------
#define LP 40   // LDS row pitch in bf16 elements (80B: conflict-free, 16B aligned)

__global__ __launch_bounds__(256) void gemm_qkv(
    const bf16_t* __restrict__ A,    // [8192][1024]
    const bf16_t* __restrict__ Wt,   // [3072][1024]
    bf16_t* __restrict__ Qb, bf16_t* __restrict__ Kb, bf16_t* __restrict__ Vt) {
  __shared__ alignas(16) bf16_t As[128 * LP];
  __shared__ alignas(16) bf16_t Bs[64 * LP];
  const int tid = threadIdx.x;
  const int lane = tid & 31, wid = tid >> 5;
  const int lane15 = lane & 15, hf = lane >> 4;
  const int waveM = wid >> 1, waveN = wid & 1;
  const int mblk = blockIdx.x * 128;
  const int nblk = blockIdx.y * 64;

  v8f acc[2][2];
  acc[0][0] = vzero8(); acc[0][1] = vzero8();
  acc[1][0] = vzero8(); acc[1][1] = vzero8();

  const int ra0 = tid >> 2,         sa0 = tid & 3;          // A chunk 0
  const int ra1 = (tid + 256) >> 2, sa1 = tid & 3;          // A chunk 1
  const int rb  = tid >> 2,         sb  = tid & 3;          // B chunk

  for (int kt = 0; kt < D_; kt += 32) {
    *(Chunk8*)&As[ra0 * LP + sa0 * 8] = *(const Chunk8*)&A[(mblk + ra0) * D_ + kt + sa0 * 8];
    *(Chunk8*)&As[ra1 * LP + sa1 * 8] = *(const Chunk8*)&A[(mblk + ra1) * D_ + kt + sa1 * 8];
    *(Chunk8*)&Bs[rb  * LP + sb  * 8] = *(const Chunk8*)&Wt[(nblk + rb) * D_ + kt + sb * 8];
    if (kt + 32 < D_) {                      // lowers to global_prefetch_b8 (GL2)
      __builtin_prefetch(&A[(mblk + ra0) * D_ + kt + 32 + sa0 * 8], 0, 3);
      __builtin_prefetch(&A[(mblk + ra1) * D_ + kt + 32 + sa1 * 8], 0, 3);
      __builtin_prefetch(&Wt[(nblk + rb) * D_ + kt + 32 + sb * 8], 0, 3);
    }
    __syncthreads();

    Frag16 af[2], bfr[2];
#pragma unroll
    for (int mi = 0; mi < 2; ++mi) {
      const bf16_t* p = &As[(waveM * 32 + mi * 16 + lane15) * LP + hf * 8];
      af[mi].q[0] = *(const Chunk8*)p;
      af[mi].q[1] = *(const Chunk8*)(p + 16);
    }
#pragma unroll
    for (int ni = 0; ni < 2; ++ni) {
      const bf16_t* p = &Bs[(waveN * 32 + ni * 16 + lane15) * LP + hf * 8];
      bfr[ni].q[0] = *(const Chunk8*)p;
      bfr[ni].q[1] = *(const Chunk8*)(p + 16);
    }
#pragma unroll
    for (int mi = 0; mi < 2; ++mi)
#pragma unroll
      for (int ni = 0; ni < 2; ++ni)
        acc[mi][ni] = wmma_bf16(af[mi], bfr[ni], acc[mi][ni]);
    __syncthreads();
  }

  // Scatter: col n -> {q|k|v}, head h, hd;  row m -> (b, s)
#pragma unroll
  for (int mi = 0; mi < 2; ++mi)
#pragma unroll
    for (int ni = 0; ni < 2; ++ni) {
      const int nfb   = nblk + waveN * 32 + ni * 16;   // 16-aligned fragment col base
      const int which = nfb >> 10;                     // 0=q 1=k 2=v (uniform per frag)
      const int hh    = (nfb >> 6) & 15;               // head (uniform per frag)
      const int hd    = (nfb & 63) + lane15;
      const int mrow0 = mblk + waveM * 32 + mi * 16 + hf * 8;
      const int b     = mrow0 >> 11;                   // /S_
      const int s0    = mrow0 & (S_ - 1);
      if (which == 0) {
        bf16_t* q = &Qb[(((size_t)b * H_ + hh) * S_ + s0) * HD_ + hd];
#pragma unroll
        for (int i = 0; i < 8; ++i) q[i * HD_] = f2bf(acc[mi][ni][i] * SCALE_);
      } else if (which == 1) {
        bf16_t* k = &Kb[(((size_t)b * H_ + hh) * S_ + s0) * HD_ + hd];
#pragma unroll
        for (int i = 0; i < 8; ++i) k[i * HD_] = f2bf(acc[mi][ni][i]);
      } else {
        union { Chunk8 c; unsigned short u[8]; } pk;
#pragma unroll
        for (int i = 0; i < 8; ++i) pk.u[i] = bfbits(acc[mi][ni][i]);
        *(Chunk8*)&Vt[(((size_t)b * H_ + hh) * HD_ + hd) * S_ + s0] = pk.c;  // 8 consec s
      }
    }
}

// ---------------------------------------------------------------------------
// Flash attention: one wave per 16-query tile, key blocks of 32.
// Q pre-scaled; V pre-transposed. Online softmax, C->A transpose via per-wave LDS.
// ---------------------------------------------------------------------------
#define AW 4
__global__ __launch_bounds__(32 * AW) void attn_fwd(
    const bf16_t* __restrict__ Qb, const bf16_t* __restrict__ Kb,
    const bf16_t* __restrict__ Vt, bf16_t* __restrict__ AO) {
  __shared__ alignas(16) bf16_t Pl[AW][16 * LP];
  const int tid = threadIdx.x, lane = tid & 31, wid = tid >> 5;
  const int lane15 = lane & 15, hf = lane >> 4;
  const int bh = blockIdx.x;                  // b*H + h
  const int b = bh >> 4, h = bh & 15;
  const int q0 = (blockIdx.y * AW + wid) * 16;
  const bf16_t* Qp = Qb + (size_t)bh * S_ * HD_;
  const bf16_t* Kp = Kb + (size_t)bh * S_ * HD_;
  const bf16_t* Vp = Vt + (size_t)bh * HD_ * S_;
  bf16_t* Pw = &Pl[wid][0];

  Frag16 qf[2];
#pragma unroll
  for (int kk = 0; kk < 2; ++kk) {
    const bf16_t* p = Qp + (q0 + lane15) * HD_ + kk * 32 + hf * 8;
    qf[kk].q[0] = *(const Chunk8*)p;
    qf[kk].q[1] = *(const Chunk8*)(p + 16);
  }

  v8f o[4];
  o[0] = vzero8(); o[1] = vzero8(); o[2] = vzero8(); o[3] = vzero8();
  float mrow[8], lrow[8];
#pragma unroll
  for (int i = 0; i < 8; ++i) { mrow[i] = -1e30f; lrow[i] = 0.f; }

  const int nkb = (q0 >> 5) + 1;              // key blocks of 32 covering causal span
  for (int kb = 0; kb < nkb; ++kb) {
    const int key0 = kb * 32;
    v8f sc[2];
    sc[0] = vzero8(); sc[1] = vzero8();
#pragma unroll
    for (int ni = 0; ni < 2; ++ni)
#pragma unroll
      for (int kk = 0; kk < 2; ++kk) {
        Frag16 kf;
        const bf16_t* p = Kp + (key0 + ni * 16 + lane15) * HD_ + kk * 32 + hf * 8;
        kf.q[0] = *(const Chunk8*)p;
        kf.q[1] = *(const Chunk8*)(p + 16);
        sc[ni] = wmma_bf16(qf[kk], kf, sc[ni]);
      }

    float p0[8], p1[8];
#pragma unroll
    for (int i = 0; i < 8; ++i) {
      const int qrow = q0 + i + hf * 8;                       // C-layout row
      float s0v = (key0 + lane15      > qrow) ? -1e30f : sc[0][i];
      float s1v = (key0 + 16 + lane15 > qrow) ? -1e30f : sc[1][i];
      float tmax = fmaxf(s0v, s1v);
#pragma unroll
      for (int off = 1; off < 16; off <<= 1)
        tmax = fmaxf(tmax, __shfl_xor(tmax, off, 32));        // reduce within 16-lane half
      const float mnew = fmaxf(mrow[i], tmax);
      const float corr = __expf(mrow[i] - mnew);
      const float e0 = __expf(s0v - mnew);
      const float e1 = __expf(s1v - mnew);
      float rs = e0 + e1;
#pragma unroll
      for (int off = 1; off < 16; off <<= 1)
        rs += __shfl_xor(rs, off, 32);
      lrow[i] = lrow[i] * corr + rs;
      mrow[i] = mnew;
      p0[i] = e0; p1[i] = e1;
#pragma unroll
      for (int f = 0; f < 4; ++f) o[f][i] *= corr;
    }

    // C-layout -> A-layout transpose of P through per-wave LDS (same-wave DS order)
#pragma unroll
    for (int i = 0; i < 8; ++i) {
      const int r = i + hf * 8;
      Pw[r * LP + lane15]      = f2bf(p0[i]);
      Pw[r * LP + 16 + lane15] = f2bf(p1[i]);
    }
    Frag16 pa;
    {
      const bf16_t* p = Pw + lane15 * LP + hf * 8;
      pa.q[0] = *(const Chunk8*)p;
      pa.q[1] = *(const Chunk8*)(p + 16);
    }
#pragma unroll
    for (int nd = 0; nd < 4; ++nd) {
      Frag16 vf;
      const bf16_t* p = Vp + (nd * 16 + lane15) * S_ + key0 + hf * 8;
      vf.q[0] = *(const Chunk8*)p;
      vf.q[1] = *(const Chunk8*)(p + 16);
      o[nd] = wmma_bf16(pa, vf, o[nd]);
    }
  }

#pragma unroll
  for (int i = 0; i < 8; ++i) {
    const float inv = 1.0f / lrow[i];
    const size_t row = (size_t)b * S_ + q0 + i + hf * 8;
#pragma unroll
    for (int nd = 0; nd < 4; ++nd)
      AO[row * D_ + h * HD_ + nd * 16 + lane15] = f2bf(o[nd][i] * inv);
  }
}

// ---------------------------------------------------------------------------
// GEMM 2: out = attn_out @ w_out  (M=8192, N=1024, K=1024), f32 output
// ---------------------------------------------------------------------------
__global__ __launch_bounds__(256) void gemm_out(
    const bf16_t* __restrict__ A,    // [8192][1024]
    const bf16_t* __restrict__ Wt,   // [1024][1024] (transposed weight)
    float* __restrict__ Out) {
  __shared__ alignas(16) bf16_t As[128 * LP];
  __shared__ alignas(16) bf16_t Bs[64 * LP];
  const int tid = threadIdx.x;
  const int lane = tid & 31, wid = tid >> 5;
  const int lane15 = lane & 15, hf = lane >> 4;
  const int waveM = wid >> 1, waveN = wid & 1;
  const int mblk = blockIdx.x * 128;
  const int nblk = blockIdx.y * 64;

  v8f acc[2][2];
  acc[0][0] = vzero8(); acc[0][1] = vzero8();
  acc[1][0] = vzero8(); acc[1][1] = vzero8();

  const int ra0 = tid >> 2, ra1 = (tid + 256) >> 2, sa = tid & 3;
  const int rb = tid >> 2, sb = tid & 3;

  for (int kt = 0; kt < D_; kt += 32) {
    *(Chunk8*)&As[ra0 * LP + sa * 8] = *(const Chunk8*)&A[(mblk + ra0) * D_ + kt + sa * 8];
    *(Chunk8*)&As[ra1 * LP + sa * 8] = *(const Chunk8*)&A[(mblk + ra1) * D_ + kt + sa * 8];
    *(Chunk8*)&Bs[rb  * LP + sb * 8] = *(const Chunk8*)&Wt[(nblk + rb) * D_ + kt + sb * 8];
    if (kt + 32 < D_) {
      __builtin_prefetch(&A[(mblk + ra0) * D_ + kt + 32 + sa * 8], 0, 3);
      __builtin_prefetch(&Wt[(nblk + rb) * D_ + kt + 32 + sb * 8], 0, 3);
    }
    __syncthreads();

    Frag16 af[2], bfr[2];
#pragma unroll
    for (int mi = 0; mi < 2; ++mi) {
      const bf16_t* p = &As[(waveM * 32 + mi * 16 + lane15) * LP + hf * 8];
      af[mi].q[0] = *(const Chunk8*)p;
      af[mi].q[1] = *(const Chunk8*)(p + 16);
    }
#pragma unroll
    for (int ni = 0; ni < 2; ++ni) {
      const bf16_t* p = &Bs[(waveN * 32 + ni * 16 + lane15) * LP + hf * 8];
      bfr[ni].q[0] = *(const Chunk8*)p;
      bfr[ni].q[1] = *(const Chunk8*)(p + 16);
    }
#pragma unroll
    for (int mi = 0; mi < 2; ++mi)
#pragma unroll
      for (int ni = 0; ni < 2; ++ni)
        acc[mi][ni] = wmma_bf16(af[mi], bfr[ni], acc[mi][ni]);
    __syncthreads();
  }

#pragma unroll
  for (int mi = 0; mi < 2; ++mi)
#pragma unroll
    for (int ni = 0; ni < 2; ++ni) {
      const int ncol  = nblk + waveN * 32 + ni * 16 + lane15;
      const int mrow0 = mblk + waveM * 32 + mi * 16 + hf * 8;
#pragma unroll
      for (int i = 0; i < 8; ++i)
        Out[(size_t)(mrow0 + i) * D_ + ncol] = acc[mi][ni][i];
    }
}

// ---------------------------------------------------------------------------
extern "C" void kernel_launch(void* const* d_in, const int* in_sizes, int n_in,
                              void* d_out, int out_size, void* d_ws, size_t ws_size,
                              hipStream_t stream) {
  (void)in_sizes; (void)n_in; (void)out_size; (void)ws_size;
  const float* x     = (const float*)d_in[0];   // [B,S,D]
  const float* w_qkv = (const float*)d_in[1];   // [D,3D]
  const float* w_out = (const float*)d_in[2];   // [D,D]
  float* out = (float*)d_out;                   // [B,S,D]

  char* ws = (char*)d_ws;
  size_t off = 0;
  const size_t XSZ = (size_t)B_ * S_ * D_;      // 8,388,608
  bf16_t* xb    = (bf16_t*)(ws + off); off += XSZ * 2;
  bf16_t* wqkvt = (bf16_t*)(ws + off); off += (size_t)D_ * N3_ * 2;
  bf16_t* woutt = (bf16_t*)(ws + off); off += (size_t)D_ * D_ * 2;
  bf16_t* Qb    = (bf16_t*)(ws + off); off += XSZ * 2;
  bf16_t* Kb    = (bf16_t*)(ws + off); off += XSZ * 2;
  bf16_t* Vt    = (bf16_t*)(ws + off); off += XSZ * 2;
  bf16_t* AO    = (bf16_t*)(ws + off); off += XSZ * 2;   // ~88 MB total

  cvt_bf16<<<(int)(XSZ / 256), 256, 0, stream>>>(x, xb, (int)XSZ);
  transpose_bf16<<<(D_ * N3_) / 256, 256, 0, stream>>>(w_qkv, wqkvt, D_, N3_);
  transpose_bf16<<<(D_ * D_) / 256, 256, 0, stream>>>(w_out, woutt, D_, D_);

  dim3 g1((B_ * S_) / 128, N3_ / 64);           // 64 x 48
  gemm_qkv<<<g1, 256, 0, stream>>>(xb, wqkvt, Qb, Kb, Vt);

  dim3 g2(B_ * H_, S_ / (16 * AW));             // 64 x 32
  attn_fwd<<<g2, 32 * AW, 0, stream>>>(Qb, Kb, Vt, AO);

  dim3 g3((B_ * S_) / 128, D_ / 64);            // 64 x 16
  gemm_out<<<g3, 256, 0, stream>>>(AO, woutt, out);
}
